// SelfAttention_44375602103173
// MI455X (gfx1250) — matrix operand order repaired and checked by reading, not compiled, not run on hardware
//
#include <hip/hip_runtime.h>
#include <hip/hip_bf16.h>

// ---------------------------------------------------------------------------
// Channel attention for MI455X (gfx1250), wave32 + v_wmma_f32_16x16x32_f16.
//
// Shapes: B=16, C=512, H=W=64 -> N=4096 pixels/batch, heads h=8, d=64.
//   pass1:  q/k/v = W*x + b       (3x [512x512]x[512x65536] GEMM, f16 WMMA)
//   pass2a: partial scores q k^T  per (bh, k-slice of 512): 1024 WGs
//   pass2b: sum partials, /8, row-softmax -> attn f16 [bh][64][64]
//   pass3:  out = attn @ v        per (bh, n-slice of 512): 1024 WGs
//
// WMMA f16 fragment layouts (CDNA5 ISA 7.12.2):
//   A (16x32, MxK): lane r=lane&15 holds row M=r; hi=lane>>4;
//       halves 0..7  = K[hi*8 .. +7], halves 8..15 = K[16+hi*8 .. +7]
//   B (32x16, KxN): lane r holds col N=r; K order as A; code loads A and B as t = K[t + 16*hi]
//   C/D (16x16 f32): vgpr j, lane -> (M = j + 8*hi, N = r)
// ---------------------------------------------------------------------------

typedef __attribute__((ext_vector_type(16))) _Float16 v16h;
typedef __attribute__((ext_vector_type(8)))  _Float16 v8h;
typedef __attribute__((ext_vector_type(8)))  float    v8f;

#define CDIM   512
#define NPIX   4096                     // pixels per batch
#define MAT_ELEMS (16ull * CDIM * NPIX) // elements of one q/k/v tensor

// A fragment: base points at A[0][k0] of a row-major [M][lda] f16 matrix.
__device__ __forceinline__ v16h load_a_frag(const _Float16* base, int lda) {
  int lane = threadIdx.x & 31;
  int r = lane & 15, hi = lane >> 4;
  const _Float16* row = base + (size_t)r * lda + hi * 16;
  v8h lo  = *(const v8h*)(row);        // K = hi*16 .. +7
  v8h hi8 = *(const v8h*)(row + 8);    // K = hi*16+8 .. +7
  v16h a;
#pragma unroll
  for (int i = 0; i < 8; ++i) { a[i] = lo[i]; a[i + 8] = hi8[i]; }
  return a;
}

// B fragment: baseT points at B^T[0][k0] of a row-major [N][ldb] f16 matrix.
__device__ __forceinline__ v16h load_b_frag(const _Float16* baseT, int ldb) {
  int lane = threadIdx.x & 31;
  int r = lane & 15, hi = lane >> 4;
  const _Float16* row = baseT + (size_t)r * ldb + hi * 16; // K = 16*hi .. +15
  return *(const v16h*)(row);
}

__device__ __forceinline__ v8f wmma_f16(v16h a, v16h b, v8f c) {
  return __builtin_amdgcn_wmma_f32_16x16x32_f16(false, a, false, b,
                                                (short)0, c, false, false);
}

// ---------------------------------------------------------------------------
// Kernel 0: convert the three fp32 512x512 weight matrices to f16 (row-major).
// ---------------------------------------------------------------------------
__global__ void convert_w(const float* __restrict__ wq,
                          const float* __restrict__ wk,
                          const float* __restrict__ wv,
                          _Float16* __restrict__ wh) {
  int i = blockIdx.x * 256 + threadIdx.x;
  if (i < CDIM * CDIM) {
    wh[i]                   = (_Float16)wq[i];
    wh[CDIM * CDIM + i]     = (_Float16)wk[i];
    wh[2 * CDIM * CDIM + i] = (_Float16)wv[i];
  }
}

// ---------------------------------------------------------------------------
// Kernel 1: fused Q/K/V projection.
// Grid: (65536/64, 512/128). Block: 256 (8 waves).
// WG tile: 128 out-channels x 64 pixels; wave w owns rows [16w,16w+16).
// K loop over C=512 in steps of 32, x-tile staged (transposed, f16) in LDS,
// double-buffered; next tile prefetched (branchless) via global_prefetch_b8.
// All three weight A-fragments are issued as one load clause before the
// 12 WMMAs so their L2 latency overlaps the matrix pipe.
// ---------------------------------------------------------------------------
__global__ __launch_bounds__(256)
void qkv_project(const float* __restrict__ x,
                 const float* __restrict__ bq,
                 const float* __restrict__ bk,
                 const float* __restrict__ bv,
                 const _Float16* __restrict__ wh,   // [3][512][512] f16
                 _Float16* __restrict__ qkv) {      // [3][16][512][4096] f16
  __shared__ alignas(16) _Float16 xT[2][64][40];    // [pixel][k], pad->40

  const int tid  = threadIdx.x;
  const int wave = tid >> 5, lane = tid & 31;
  const int r = lane & 15, hi = lane >> 4;

  const int p0 = blockIdx.x * 64;          // first pixel of tile (64 | 4096)
  const int b  = p0 >> 12;                 // batch
  const int n0 = p0 & (NPIX - 1);          // pixel within batch
  const int m0 = blockIdx.y * 128 + wave * 16;   // out-channel row base

  const float* xb = x + (size_t)(b * CDIM) * NPIX + n0;

  v8f acc[3][4];
#pragma unroll
  for (int m = 0; m < 3; ++m)
#pragma unroll
    for (int t = 0; t < 4; ++t) acc[m][t] = v8f{};

  const int lrow = tid >> 3;        // 0..31  (k row of the staged tile)
  const int lcol = (tid & 7) * 8;   // 0..56  (pixel base, 8 per thread)

  for (int kc = 0; kc < CDIM; kc += 32) {
    const int buf = (kc >> 5) & 1;
    // stage x[kc+lrow][n0+lcol..+8) -> xT[buf][pixel][k]  (fp32 -> f16)
    const float* src = xb + (size_t)(kc + lrow) * NPIX + lcol;
    float4 f0 = *(const float4*)(src);
    float4 f1 = *(const float4*)(src + 4);
    // branchless prefetch of next tile's rows (clamped on last iteration)
    const float* pf = src + (kc < CDIM - 32 ? 32 * NPIX : 0);
    __builtin_prefetch(pf, 0, 1);
    _Float16 h8[8] = {(_Float16)f0.x, (_Float16)f0.y, (_Float16)f0.z,
                      (_Float16)f0.w, (_Float16)f1.x, (_Float16)f1.y,
                      (_Float16)f1.z, (_Float16)f1.w};
#pragma unroll
    for (int j = 0; j < 8; ++j) xT[buf][lcol + j][lrow] = h8[j];
    __syncthreads();

    // issue all weight fragments first (one clause), then the 12 WMMAs
    v16h afr[3];
#pragma unroll
    for (int m = 0; m < 3; ++m)
      afr[m] = load_a_frag(wh + ((size_t)m * CDIM + m0) * CDIM + kc, CDIM);
    v16h bfr[4];
#pragma unroll
    for (int t = 0; t < 4; ++t) bfr[t] = load_b_frag(&xT[buf][t * 16][0], 40);
#pragma unroll
    for (int m = 0; m < 3; ++m)
#pragma unroll
      for (int t = 0; t < 4; ++t) acc[m][t] = wmma_f16(afr[m], bfr[t], acc[m][t]);
    // double-buffered: next iteration stores into the other buffer
  }

  const float* biases[3] = {bq, bk, bv};
#pragma unroll
  for (int m = 0; m < 3; ++m) {
    _Float16* outm = qkv + (size_t)m * MAT_ELEMS +
                     ((size_t)(b * CDIM + m0)) * NPIX + n0;
#pragma unroll
    for (int j = 0; j < 8; ++j) {
      const int row = j + 8 * hi;
      const float bb = biases[m][m0 + row];
#pragma unroll
      for (int t = 0; t < 4; ++t)
        outm[(size_t)row * NPIX + t * 16 + r] = (_Float16)(acc[m][t][j] + bb);
    }
  }
}

// ---------------------------------------------------------------------------
// Kernel 2a: partial channel-Gram scores.
// Grid: (8 k-slices, 128 bh). Block: 256 (8 waves).
// Wave w: mi=w&3 -> rows [16mi,+16); ci=w>>2 -> cols [32ci,+32).
// Each WG reduces K in [512*slice, +512) and writes a 64x64 f32 partial.
// ---------------------------------------------------------------------------
__global__ __launch_bounds__(256)
void attn_scores(const _Float16* __restrict__ qkv,
                 float* __restrict__ sp) {          // [128][8][64][64] f32
  const int tid  = threadIdx.x;
  const int wave = tid >> 5, lane = tid & 31;
  const int r = lane & 15, hi = lane >> 4;

  const int slice = blockIdx.x;               // 0..7
  const int bh    = blockIdx.y;               // 0..127
  const int b = bh >> 3, h = bh & 7;
  const size_t base = ((size_t)(b * CDIM + h * 64)) * NPIX;
  const _Float16* qb = qkv + base;
  const _Float16* kb = qkv + MAT_ELEMS + base;

  const int mi = wave & 3, ci = wave >> 2;
  const int m0 = mi * 16;
  const int k0 = slice * 512;

  v8f s0 = v8f{}, s1 = v8f{};
  for (int k = k0; k < k0 + 512; k += 32) {
    v16h af = load_a_frag(qb + (size_t)m0 * NPIX + k, NPIX);
    v16h b0 = load_b_frag(kb + (size_t)(ci * 32) * NPIX + k, NPIX);
    v16h b1 = load_b_frag(kb + (size_t)(ci * 32 + 16) * NPIX + k, NPIX);
    s0 = wmma_f16(af, b0, s0);
    s1 = wmma_f16(af, b1, s1);
  }
  float* spb = sp + ((size_t)bh * 8 + slice) * 4096;
#pragma unroll
  for (int j = 0; j < 8; ++j) {
    spb[(m0 + j + 8 * hi) * 64 + ci * 32 + r]      = s0[j];
    spb[(m0 + j + 8 * hi) * 64 + ci * 32 + 16 + r] = s1[j];
  }
}

// ---------------------------------------------------------------------------
// Kernel 2b: reduce partials, scale by 1/8, row-softmax, emit attn f16.
// Grid: 128. Block: 256.
// ---------------------------------------------------------------------------
__global__ __launch_bounds__(256)
void attn_softmax(const float* __restrict__ sp,    // [128][8][64][64]
                  _Float16* __restrict__ attn_g) { // [128][64][64]
  __shared__ float sc[64][68];
  const int tid = threadIdx.x;
  const int bh  = blockIdx.x;
  const float* spb = sp + (size_t)bh * 8 * 4096;

  const int row = tid >> 2;          // 0..63
  const int e0  = (tid & 3) * 16;    // 0..48
  for (int e = e0; e < e0 + 16; ++e) {
    float s = 0.f;
#pragma unroll
    for (int sl = 0; sl < 8; ++sl) s += spb[sl * 4096 + row * 64 + e];
    sc[row][e] = s * 0.125f;
  }
  __syncthreads();

  if (tid < 64) {
    float mx = -3.4e38f;
    for (int e = 0; e < 64; ++e) mx = fmaxf(mx, sc[tid][e]);
    float sum = 0.f;
    for (int e = 0; e < 64; ++e) {
      float v = __expf(sc[tid][e] - mx);
      sc[tid][e] = v;
      sum += v;
    }
    float inv = 1.f / sum;
    _Float16* arow = attn_g + (size_t)bh * 4096 + tid * 64;
    for (int e = 0; e < 64; ++e) arow[e] = (_Float16)(sc[tid][e] * inv);
  }
}

// ---------------------------------------------------------------------------
// Kernel 3: out = attn @ v.
// Grid: (8 n-slices, 128 bh). Block: 256 (8 waves).
// Each WG covers 512 output columns in 8 chunks of 64; v chunk is transposed
// into double-buffered LDS so B-fragments are aligned ds_load_b128.
// Attention A-fragments are loop-invariant (hoisted, from global f16).
// ---------------------------------------------------------------------------
__global__ __launch_bounds__(256)
void attn_out(const _Float16* __restrict__ qkv,
              const _Float16* __restrict__ attn_g,
              float* __restrict__ out) {
  __shared__ alignas(16) _Float16 vT[2][64][80];   // v chunk, [n][e]

  const int tid  = threadIdx.x;
  const int wave = tid >> 5, lane = tid & 31;
  const int r = lane & 15, hi = lane >> 4;

  const int bh = blockIdx.y;                  // 0..127
  const int b = bh >> 3, h = bh & 7;
  const size_t base = ((size_t)(b * CDIM + h * 64)) * NPIX;
  const _Float16* vb = qkv + 2 * MAT_ELEMS + base;
  const int n_base = blockIdx.x * 512;

  const int mi = wave & 3, ci = wave >> 2;
  const int m0 = mi * 16;

  // attn A-fragments: rows m0..m0+16, K (=e) chunks 0 and 32; lda = 64
  const _Float16* ab = attn_g + (size_t)bh * 4096 + m0 * 64;
  const v16h af0 = load_a_frag(ab, 64);
  const v16h af1 = load_a_frag(ab + 32, 64);

  const int erow = tid >> 2;           // 0..63 : e row of staged v chunk
  const int cg   = (tid & 3) * 16;     // 0..48 : column base (16 per thread)
  for (int cc = 0; cc < 512; cc += 64) {
    const int c64 = n_base + cc;
    const int buf = (cc >> 6) & 1;
    const _Float16* src = vb + (size_t)erow * NPIX + c64 + cg;
    v8h p0 = *(const v8h*)(src);
    v8h p1 = *(const v8h*)(src + 8);
#pragma unroll
    for (int j = 0; j < 8; ++j) {      // transpose into [n][e]
      vT[buf][cg + j][erow]     = p0[j];
      vT[buf][cg + 8 + j][erow] = p1[j];
    }
    __syncthreads();

    v8f o0 = v8f{}, o1 = v8f{};
    {
      v16h bf0 = load_b_frag(&vT[buf][ci * 32][0], 80);
      v16h bf1 = load_b_frag(&vT[buf][ci * 32 + 16][0], 80);
      o0 = wmma_f16(af0, bf0, o0);
      o1 = wmma_f16(af0, bf1, o1);
      bf0 = load_b_frag(&vT[buf][ci * 32][32], 80);
      bf1 = load_b_frag(&vT[buf][ci * 32 + 16][32], 80);
      o0 = wmma_f16(af1, bf0, o0);
      o1 = wmma_f16(af1, bf1, o1);
    }
    float* ob = out + base + c64;
#pragma unroll
    for (int j = 0; j < 8; ++j) {
      ob[(size_t)(m0 + j + 8 * hi) * NPIX + ci * 32 + r]      = o0[j];
      ob[(size_t)(m0 + j + 8 * hi) * NPIX + ci * 32 + 16 + r] = o1[j];
    }
    // double-buffered vT: next chunk stores into the other buffer
  }
}

// ---------------------------------------------------------------------------
extern "C" void kernel_launch(void* const* d_in, const int* in_sizes, int n_in,
                              void* d_out, int out_size, void* d_ws,
                              size_t ws_size, hipStream_t stream) {
  (void)in_sizes; (void)n_in; (void)out_size; (void)ws_size;
  const float* x  = (const float*)d_in[0];
  const float* wq = (const float*)d_in[1];
  const float* bq = (const float*)d_in[2];
  const float* wk = (const float*)d_in[3];
  const float* bk = (const float*)d_in[4];
  const float* wv = (const float*)d_in[5];
  const float* bv = (const float*)d_in[6];
  float* out = (float*)d_out;

  // workspace layout: [q f16][k f16][v f16][weights f16][sp f32][attn f16]
  _Float16* qkv = (_Float16*)d_ws;
  _Float16* wh  = qkv + 3 * MAT_ELEMS;                 // 786432 halves
  float*    sp  = (float*)(wh + 3 * CDIM * CDIM);      // 128*8*64*64 f32
  _Float16* attn_g = (_Float16*)(sp + 128ull * 8 * 4096);  // 128*64*64 f16

  convert_w<<<(CDIM * CDIM + 255) / 256, 256, 0, stream>>>(wq, wk, wv, wh);
  qkv_project<<<dim3(65536 / 64, CDIM / 128), 256, 0, stream>>>(
      x, bq, bk, bv, wh, qkv);
  attn_scores<<<dim3(8, 128), 256, 0, stream>>>(qkv, sp);
  attn_softmax<<<128, 256, 0, stream>>>(sp, attn_g);
  attn_out<<<dim3(8, 128), 256, 0, stream>>>(qkv, attn_g, out);
}
